// ImprovedGCN_59081570124166
// MI455X (gfx1250) — matrix operand order repaired and checked by reading, not compiled, not run on hardware
//
#include <hip/hip_runtime.h>
#include <hip/hip_bf16.h>
#include <stddef.h>

// ---------------- problem constants ----------------
#define NNODES  50000
#define NEDGES  800000
#define NGRAPHS 64
#define MPAD    50048          // 50000 rounded up to 128 (GEMM M block tile)
#define F_IN    128
#define H1      512
#define H2      256
#define H3      128
#define NCLS    10
#define BN_EPS  1e-5f

typedef __bf16 bf16_t;
typedef __attribute__((ext_vector_type(16))) __bf16 v16bf;
typedef __attribute__((ext_vector_type(8)))  __bf16 v8bf;
typedef __attribute__((ext_vector_type(8)))  float  v8f;
typedef int v4i_t __attribute__((__vector_size__(4 * sizeof(int))));  // payload type of async b128

__device__ __forceinline__ v16bf cat16(v8bf lo, v8bf hi) {
  return __builtin_shufflevector(lo, hi, 0,1,2,3,4,5,6,7,8,9,10,11,12,13,14,15);
}

// -------- CDNA5 async global->LDS staging (ASYNCcnt path), with safe fallback --------
#if defined(__HIP_DEVICE_COMPILE__) && __has_builtin(__builtin_amdgcn_global_load_async_to_lds_b128)
#define GCN_ASYNC 1
#else
#define GCN_ASYNC 0
#endif

__device__ __forceinline__ void stage16(const bf16_t* g, bf16_t* l) {
#if GCN_ASYNC
  // 16B per lane, memory -> LDS directly, tracked by ASYNCcnt
  __builtin_amdgcn_global_load_async_to_lds_b128(
      (__attribute__((address_space(1))) v4i_t*)g,
      (__attribute__((address_space(3))) v4i_t*)l, 0, 0);
#else
  *(v8bf*)l = *(const v8bf*)g;      // global_load_b128 + ds_store_b128
#endif
}

__device__ __forceinline__ void stage_wait_and_barrier() {
#if GCN_ASYNC
#if __has_builtin(__builtin_amdgcn_s_wait_asynccnt)
  __builtin_amdgcn_s_wait_asynccnt(0);
#else
  asm volatile("s_wait_asynccnt 0" ::: "memory");
#endif
#endif
  __syncthreads();
}

// ---------------- WMMA bf16 GEMM:  C[MPAD x N] = A[MPAD x K] * W  (W given as W^T [N x K]) ----
// block = 256 threads = 8 waves; block tile 128(M) x 64(N); wave tile 16(M) x 64(N).
// B tile (64 cols x 32 K, 4KB) staged once per block per K-step into double-buffered LDS.
#define BROW 40   // padded LDS row stride in halves (80B: conflict-free, 16B-aligned)

__global__ __launch_bounds__(256)
void gemm_wmma_bf16(const bf16_t* __restrict__ A,   // MPAD x K, row-major
                    const bf16_t* __restrict__ WT,  // N x K, row-major (transposed weights)
                    float* __restrict__ C,          // MPAD x N
                    int K, int N) {
  __shared__ __align__(16) bf16_t btile[2][64 * BROW];

  const int t    = threadIdx.x;
  const int lane = t & 31;
  const int wave = t >> 5;
  const int g    = lane >> 4;    // lane half-group: selects K sub-range
  const int ln   = lane & 15;    // M index (A) / N index (B,C)
  const int m0   = blockIdx.y * 128 + wave * 16;
  const int nb   = blockIdx.x * 64;

  // staging map: thread t copies 16B: LDS col = t/4, K chunk = (t%4)*8 halves
  const int scol = t >> 2;
  const int skk  = (t & 3) * 8;
  const bf16_t* gstage = WT + (size_t)(nb + scol) * K + skk;
  const int     lstage = scol * BROW + skk;

  v8f acc[4] = {v8f{}, v8f{}, v8f{}, v8f{}};
  const bf16_t* arow = A + (size_t)(m0 + ln) * K;

  const int nsteps = K >> 5;
  stage16(gstage, &btile[0][lstage]);            // prefetch first B tile
  for (int s = 0; s < nsteps; ++s) {
    const int k0  = s << 5;
    const int buf = s & 1;
    stage_wait_and_barrier();                    // tile[buf] ready for all waves
    if (s + 1 < nsteps)                          // prefetch next tile (uniform branch)
      stage16(gstage + k0 + 32, &btile[buf ^ 1][lstage]);

    // A fragment: halves 0..7 = K[k0+8g .. +7], halves 8..15 = K[k0+16+8g .. +7]
    v8bf a_lo = *(const v8bf*)(arow + k0 + 8 * g);
    v8bf a_hi = *(const v8bf*)(arow + k0 + 16 + 8 * g);
    v16bf afrag = cat16(a_lo, a_hi);
#pragma unroll
    for (int nt = 0; nt < 4; ++nt) {
      // B fragment from LDS: lane group g holds K[8g .. 8g+7] and K[16+8g .. 16+8g+7] of column nt*16+ln
      const bf16_t* bl = &btile[buf][(nt * 16 + ln) * BROW];
      v16bf bfrag = cat16(*(const v8bf*)(bl + 8 * g), *(const v8bf*)(bl + 16 + 8 * g));
      acc[nt] = __builtin_amdgcn_wmma_f32_16x16x32_bf16(
          /*neg_a=*/false, afrag, /*neg_b=*/false, bfrag,
          /*c_mod=*/(short)0, acc[nt], /*reuse_a=*/false, /*reuse_b=*/false);
    }
  }
  // C layout: VGPR r -> row m0 + r + 8g, col nb + nt*16 + ln
#pragma unroll
  for (int nt = 0; nt < 4; ++nt) {
    float* crow = C + (size_t)(m0 + 8 * g) * N + nb + nt * 16 + ln;
#pragma unroll
    for (int r = 0; r < 8; ++r) crow[(size_t)r * N] = acc[nt][r];
  }
}

// ---------------- small utility kernels ----------------
__global__ void fill_f32_k(float* p, int n, float v) {
  int i = blockIdx.x * blockDim.x + threadIdx.x;
  if (i < n) p[i] = v;
}

__global__ void deg_edges_k(const int* __restrict__ col, float* __restrict__ deg) {
  int e = blockIdx.x * blockDim.x + threadIdx.x;
  if (e < NEDGES) atomicAdd(&deg[col[e]], 1.0f);
}

__global__ void dinv_k(const float* __restrict__ deg, float* __restrict__ dinv) {
  int i = blockIdx.x * blockDim.x + threadIdx.x;
  if (i < NNODES) dinv[i] = rsqrtf(deg[i]);   // deg >= 1 (self loop)
}

__global__ void conv_x_k(const float* __restrict__ x, bf16_t* __restrict__ IN) {
  long idx = (long)blockIdx.x * blockDim.x + threadIdx.x;
  if (idx >= (long)MPAD * F_IN) return;
  int i = (int)(idx >> 7);                    // F_IN == 128
  float v = (i < NNODES) ? x[idx] : 0.0f;
  IN[idx] = (bf16_t)v;
}

__global__ void conv_wt_k(const float* __restrict__ W, bf16_t* __restrict__ WT, int K, int N) {
  int idx = blockIdx.x * blockDim.x + threadIdx.x;
  if (idx >= K * N) return;
  int n = idx / K, k = idx % K;
  WT[idx] = (bf16_t)W[(size_t)k * N + n];     // WT[n][k] = W[k][n]
}

// agg[i] = h[i] * dinv[i]^2 + bias  (self-loop term + bias), i < NNODES
template <int N>
__global__ void agg_init_k(const float* __restrict__ H, const float* __restrict__ dinv,
                           const float* __restrict__ bias, float* __restrict__ AGG) {
  long idx = (long)blockIdx.x * blockDim.x + threadIdx.x;
  if (idx >= (long)NNODES * N) return;
  int i = (int)(idx / N), f = (int)(idx % N);
  float d = dinv[i];
  AGG[idx] = H[idx] * d * d + bias[f];
}

// edge scatter: agg[col] += h[row] * dinv[row]*dinv[col]; one float4 chunk per thread
template <int N>
__global__ __launch_bounds__(256)
void agg_edges_k(const int* __restrict__ row, const int* __restrict__ col,
                 const float* __restrict__ dinv, const float* __restrict__ H,
                 float* __restrict__ AGG) {
  constexpr int N4  = N / 4;
  constexpr int EPB = 256 / N4;
  int t = threadIdx.x;
  int e = blockIdx.x * EPB + t / N4;
  int c = (t % N4) * 4;
  if (e >= NEDGES) return;
  int r = row[e], d = col[e];
  float s = dinv[r] * dinv[d];
  float4 hv = *(const float4*)(H + (size_t)r * N + c);
  float* ap = AGG + (size_t)d * N + c;
  atomicAdd(ap + 0, hv.x * s);
  atomicAdd(ap + 1, hv.y * s);
  atomicAdd(ap + 2, hv.z * s);
  atomicAdd(ap + 3, hv.w * s);
}

// per-feature sum and sum-of-squares (256-row chunk per block, 1 atomic per feature per block)
template <int N>
__global__ __launch_bounds__(256)
void bn_stats_k(const float* __restrict__ AGG, float* __restrict__ bsum, float* __restrict__ bsq) {
  constexpr int NF = (N + 255) / 256;
  float s[NF], q[NF];
#pragma unroll
  for (int j = 0; j < NF; ++j) { s[j] = 0.0f; q[j] = 0.0f; }
  int r0 = blockIdx.x * 256;
  int r1 = min(r0 + 256, NNODES);
  for (int r = r0; r < r1; ++r) {
    const float* rowp = AGG + (size_t)r * N;
#pragma unroll
    for (int j = 0; j < NF; ++j) {
      int f = threadIdx.x + j * 256;
      if (f < N) { float v = rowp[f]; s[j] += v; q[j] += v * v; }
    }
  }
#pragma unroll
  for (int j = 0; j < NF; ++j) {
    int f = threadIdx.x + j * 256;
    if (f < N) { atomicAdd(&bsum[f], s[j]); atomicAdd(&bsq[f], q[j]); }
  }
}

__global__ void bn_fin_k(const float* __restrict__ bsum, const float* __restrict__ bsq,
                         const float* __restrict__ gam, const float* __restrict__ bet,
                         float* __restrict__ sc, float* __restrict__ sh, int N) {
  int f = blockIdx.x * blockDim.x + threadIdx.x;
  if (f >= N) return;
  const float inv_n = 1.0f / (float)NNODES;
  float mean = bsum[f] * inv_n;
  float var  = bsq[f] * inv_n - mean * mean;   // biased variance, matches reference
  float s    = gam[f] * rsqrtf(var + BN_EPS);
  sc[f] = s;
  sh[f] = bet[f] - mean * s;
}

// fused BN + ReLU + fp32->bf16 convert; also zero-fills GEMM pad rows
template <int N>
__global__ void bn_relu_k(const float* __restrict__ AGG, const float* __restrict__ sc,
                          const float* __restrict__ sh, bf16_t* __restrict__ OUT) {
  long idx = (long)blockIdx.x * blockDim.x + threadIdx.x;
  if (idx >= (long)MPAD * N) return;
  int i = (int)(idx / N), f = (int)(idx % N);
  float v = 0.0f;
  if (i < NNODES) v = fmaxf(AGG[idx] * sc[f] + sh[f], 0.0f);
  OUT[idx] = (bf16_t)v;
}

__global__ void pool_cnt_k(const int* __restrict__ batch, float* __restrict__ cnt) {
  int i = blockIdx.x * blockDim.x + threadIdx.x;
  if (i < NNODES) atomicAdd(&cnt[batch[i]], 1.0f);
}

__global__ void pool_sum_k(const bf16_t* __restrict__ h, const int* __restrict__ batch,
                           float* __restrict__ pooled) {
  long idx = (long)blockIdx.x * blockDim.x + threadIdx.x;
  if (idx >= (long)NNODES * H3) return;
  int i = (int)(idx >> 7);      // H3 == 128
  int f = (int)(idx & 127);
  atomicAdd(&pooled[batch[i] * H3 + f], (float)h[idx]);
}

__global__ void final_k(const float* __restrict__ pooled, const float* __restrict__ cnt,
                        const float* __restrict__ Wo, const float* __restrict__ bo,
                        float* __restrict__ out) {
  int t = blockIdx.x * blockDim.x + threadIdx.x;
  if (t >= NGRAPHS * NCLS) return;
  int g = t / NCLS, c = t % NCLS;
  float inv = 1.0f / fmaxf(cnt[g], 1.0f);
  float acc = bo[c];
  for (int k = 0; k < H3; ++k) acc += pooled[g * H3 + k] * inv * Wo[k * NCLS + c];
  out[t] = acc;
}

// ---------------- per-layer driver (template keeps feature-dim divides compile-time) ---------
template <int N>
static void run_agg_bn(const float* Hbuf, const float* dinv, const float* bias,
                       float* AGG, const int* row, const int* col,
                       float* bsum, float* bsq, float* bsc, float* bsh,
                       const float* gam, const float* bet, bf16_t* OUTbf,
                       hipStream_t stream) {
  const int B = 256;
  agg_init_k<N><<<(int)(((long)NNODES * N + B - 1) / B), B, 0, stream>>>(Hbuf, dinv, bias, AGG);
  constexpr int EPB = 1024 / N;  // edges per 256-thread block
  agg_edges_k<N><<<(NEDGES + EPB - 1) / EPB, B, 0, stream>>>(row, col, dinv, Hbuf, AGG);
  fill_f32_k<<<(N + B - 1) / B, B, 0, stream>>>(bsum, N, 0.0f);
  fill_f32_k<<<(N + B - 1) / B, B, 0, stream>>>(bsq, N, 0.0f);
  bn_stats_k<N><<<(NNODES + 255) / 256, 256, 0, stream>>>(AGG, bsum, bsq);
  bn_fin_k<<<(N + B - 1) / B, B, 0, stream>>>(bsum, bsq, gam, bet, bsc, bsh, N);
  bn_relu_k<N><<<(int)(((long)MPAD * N + B - 1) / B), B, 0, stream>>>(AGG, bsc, bsh, OUTbf);
}

// ---------------- entry point ----------------
extern "C" void kernel_launch(void* const* d_in, const int* in_sizes, int n_in,
                              void* d_out, int out_size, void* d_ws, size_t ws_size,
                              hipStream_t stream) {
  (void)in_sizes; (void)n_in; (void)out_size; (void)ws_size;
  const float* x    = (const float*)d_in[0];
  const int*   ei   = (const int*)d_in[1];
  const int*   row  = ei;             // sources
  const int*   col  = ei + NEDGES;    // targets
  const int*   batch= (const int*)d_in[2];
  const float* W[3]  = {(const float*)d_in[3],  (const float*)d_in[7],  (const float*)d_in[11]};
  const float* bia[3]= {(const float*)d_in[4],  (const float*)d_in[8],  (const float*)d_in[12]};
  const float* gam[3]= {(const float*)d_in[5],  (const float*)d_in[9],  (const float*)d_in[13]};
  const float* bet[3]= {(const float*)d_in[6],  (const float*)d_in[10], (const float*)d_in[14]};
  const float* Wo = (const float*)d_in[15];
  const float* bo = (const float*)d_in[16];
  float* out = (float*)d_out;

  // workspace carve-up (256B aligned)
  char* ws = (char*)d_ws;
  size_t off = 0;
  auto take = [&](size_t bytes) { char* p = ws + off; off += (bytes + 255) & ~(size_t)255; return p; };
  bf16_t* IN   = (bf16_t*)take((size_t)MPAD * H1 * sizeof(bf16_t)); // bf16 layer input (max 512 feats)
  float*  Hbuf = (float*) take((size_t)MPAD * H1 * sizeof(float));  // GEMM output
  float*  AGG  = (float*) take((size_t)MPAD * H1 * sizeof(float));  // aggregated output
  bf16_t* WT   = (bf16_t*)take((size_t)H1 * H2 * sizeof(bf16_t));   // transposed weights (max 512x256)
  float*  deg  = (float*) take((size_t)NNODES * sizeof(float));
  float*  dinv = (float*) take((size_t)NNODES * sizeof(float));
  float*  bsum = (float*) take((size_t)H1 * sizeof(float));
  float*  bsq  = (float*) take((size_t)H1 * sizeof(float));
  float*  bsc  = (float*) take((size_t)H1 * sizeof(float));
  float*  bsh  = (float*) take((size_t)H1 * sizeof(float));
  float*  pooled=(float*) take((size_t)NGRAPHS * H3 * sizeof(float));
  float*  cnt  = (float*) take((size_t)NGRAPHS * sizeof(float));

  const int B = 256;

  // degrees (self-loop folded in as init value 1.0) and dinv = rsqrt(deg)
  fill_f32_k<<<(NNODES + B - 1) / B, B, 0, stream>>>(deg, NNODES, 1.0f);
  deg_edges_k<<<(NEDGES + B - 1) / B, B, 0, stream>>>(col, deg);
  dinv_k<<<(NNODES + B - 1) / B, B, 0, stream>>>(deg, dinv);

  // input features -> bf16 (pad rows zero)
  conv_x_k<<<(int)(((long)MPAD * F_IN + B - 1) / B), B, 0, stream>>>(x, IN);

  // ----- layer 1: 128 -> 512 -----
  conv_wt_k<<<(F_IN * H1 + B - 1) / B, B, 0, stream>>>(W[0], WT, F_IN, H1);
  gemm_wmma_bf16<<<dim3(H1 / 64, MPAD / 128), 256, 0, stream>>>(IN, WT, Hbuf, F_IN, H1);
  run_agg_bn<H1>(Hbuf, dinv, bia[0], AGG, row, col, bsum, bsq, bsc, bsh,
                 gam[0], bet[0], IN, stream);

  // ----- layer 2: 512 -> 256 -----
  conv_wt_k<<<(H1 * H2 + B - 1) / B, B, 0, stream>>>(W[1], WT, H1, H2);
  gemm_wmma_bf16<<<dim3(H2 / 64, MPAD / 128), 256, 0, stream>>>(IN, WT, Hbuf, H1, H2);
  run_agg_bn<H2>(Hbuf, dinv, bia[1], AGG, row, col, bsum, bsq, bsc, bsh,
                 gam[1], bet[1], IN, stream);

  // ----- layer 3: 256 -> 128 -----
  conv_wt_k<<<(H2 * H3 + B - 1) / B, B, 0, stream>>>(W[2], WT, H2, H3);
  gemm_wmma_bf16<<<dim3(H3 / 64, MPAD / 128), 256, 0, stream>>>(IN, WT, Hbuf, H2, H3);
  run_agg_bn<H3>(Hbuf, dinv, bia[2], AGG, row, col, bsum, bsq, bsc, bsh,
                 gam[2], bet[2], IN, stream);

  // ----- global mean pool + classifier -----
  fill_f32_k<<<(NGRAPHS * H3 + B - 1) / B, B, 0, stream>>>(pooled, NGRAPHS * H3, 0.0f);
  fill_f32_k<<<1, 64, 0, stream>>>(cnt, NGRAPHS, 0.0f);
  pool_cnt_k<<<(NNODES + B - 1) / B, B, 0, stream>>>(batch, cnt);
  pool_sum_k<<<(int)(((long)NNODES * H3 + B - 1) / B), B, 0, stream>>>(IN, batch, pooled);
  final_k<<<(NGRAPHS * NCLS + B - 1) / B, B, 0, stream>>>(pooled, cnt, Wo, bo, out);
}
